// LBKT_20177756357262
// MI455X (gfx1250) — matrix operand order repaired and checked by reading, not compiled, not run on hardware
//
#include <hip/hip_runtime.h>

typedef __attribute__((ext_vector_type(16))) _Float16 v16h;
typedef __attribute__((ext_vector_type(8)))  float    v8f;

#define B_  64
#define L_  128
#define C_  256
#define DH_ 128
#define DF_ 50
#define R_  4

__device__ __forceinline__ float sigm(float x) {
    return __builtin_amdgcn_rcpf(1.f + __expf(-x));
}
// Layer(dim_h, d, b, k) swap: k_eff=0.7, b_eff=0.3, d=10
__device__ __forceinline__ float gatef(float f) {
    return 0.7f + 0.3f * __builtin_amdgcn_rcpf(1.f + __expf(-10.f * (f - 0.3f)));
}

// ---------------- prep kernels (run every call; deterministic) ----------------

// h[b,c,k] = init_h[c,k]
__global__ void k_init_h(float* __restrict__ h, const float* __restrict__ init_h) {
    int i = blockIdx.x * 256 + threadIdx.x;      // over C*DH = 32768
    float v = init_h[i];
    const int total = C_ * DH_;
    for (int b = 0; b < B_; ++b) h[(size_t)b * total + i] = v;
}

// Pack W1 = gate3_W[0:128,:] into WMMA B-fragment order (f16), and compute the
// column sums of the three factor blocks (rows 256..405) of gate3_W.
__global__ void k_prep_w1(_Float16* __restrict__ W1frag, float* __restrict__ s345,
                          const float* __restrict__ g3W) {
    int idx  = blockIdx.x * 256 + threadIdx.x;   // 0..16383
    int j    = idx & 15;
    int lane = (idx >> 4) & 31;
    int kt   = (idx >> 9) & 3;
    int nt   = idx >> 11;
    int K = kt * 32 + ((lane >> 4) << 4) + j;    // B layout: lanes 0-15 K=0..15, 16-31 K=16..31
    int N = nt * 16 + (lane & 15);
    W1frag[idx] = (_Float16)g3W[K * DH_ + N];
    if (idx < 3 * DH_) {
        int which = idx / DH_, hcol = idx % DH_;
        float s = 0.f;
        int row0 = 2 * DH_ + which * DF_;
        for (int r = 0; r < DF_; ++r) s += g3W[(row0 + r) * DH_ + hcol];
        s345[idx] = s;
    }
}

// acts[b,t,h] = relu([q_emb|c_emb] @ input_W + input_b)
__global__ void k_acts(float* __restrict__ acts,
                       const int* __restrict__ qseq, const int* __restrict__ cseq,
                       const float* __restrict__ eq, const float* __restrict__ ec,
                       const float* __restrict__ inW, const float* __restrict__ inb) {
    __shared__ float x[2 * DH_];
    int bt = blockIdx.x;                         // b*L + t
    int tid = threadIdx.x;                       // 0..127
    int qid = qseq[bt], cid = cseq[bt];
    x[tid]       = eq[(size_t)qid * DH_ + tid];
    x[DH_ + tid] = ec[(size_t)cid * DH_ + tid];
    __syncthreads();
    float acc = inb[tid];
    for (int k = 0; k < 2 * DH_; ++k) acc += x[k] * inW[k * DH_ + tid];
    acts[(size_t)bt * DH_ + tid] = fmaxf(acc, 0.f);
}

// Seed per-tile read partials for t=0 from init_h:
// part[b][ct][h] = sum_{c in tile ct} corr_0[b,c] * init_h[c,h]
__global__ void k_part0(float* __restrict__ part, const float* __restrict__ init_h,
                        const float* __restrict__ qmat, const int* __restrict__ qseq) {
    int ct = blockIdx.x, b = blockIdx.y, tid = threadIdx.x;   // tid = h
    int qid = qseq[b * L_ + 0];
    const float* qm = qmat + (size_t)qid * C_ + ct * 16;
    float acc = 0.f;
    for (int i = 0; i < 16; ++i) acc += qm[i] * init_h[(ct * 16 + i) * DH_ + tid];
    part[((b * 16 + ct) * DH_) + tid] = acc;
}

// ---------------- per-step kernels ----------------

// gates, prediction, learning_gain, and the C-independent part of the forget logit.
// h_tilde is reduced from the 16 per-tile partials produced by k_update/k_part0.
__global__ void k_gates(float* __restrict__ lg, float* __restrict__ g2b, float* __restrict__ out,
                        const float* __restrict__ part, const float* __restrict__ acts,
                        const int* __restrict__ qseq, const float* __restrict__ eq,
                        const float* __restrict__ tf, const float* __restrict__ af,
                        const float* __restrict__ hf,
                        const float* __restrict__ tgW, const float* __restrict__ tgb,
                        const float* __restrict__ agW, const float* __restrict__ agb,
                        const float* __restrict__ hgW, const float* __restrict__ hgb,
                        const float* __restrict__ tw,  const float* __restrict__ aw,
                        const float* __restrict__ hw,
                        const float* __restrict__ Wff, const float* __restrict__ cb,
                        const float* __restrict__ g3W, const float* __restrict__ g3b,
                        const float* __restrict__ s345,
                        const float* __restrict__ outW, const float* __restrict__ outb, int t) {
    __shared__ float sht[DH_], sx[DH_], sq[DH_], stg[DH_], sag[DH_], shg[DH_], sred[DH_];
    int b = blockIdx.x, tid = threadIdx.x;
    int bt = b * L_ + t;
    {   // h_tilde[b,tid] = sum over 16 tile partials (fixed order -> deterministic)
        float acc = 0.f;
        const float* pb = part + (size_t)b * 16 * DH_ + tid;
        for (int ct = 0; ct < 16; ++ct) acc += pb[ct * DH_];
        sht[tid] = acc;
    }
    sx[tid] = acts[(size_t)bt * DH_ + tid];
    int qid = qseq[bt];
    sq[tid] = eq[(size_t)qid * DH_ + tid];
    __syncthreads();

    float a_t = 0.f, a_a = 0.f, a_h = 0.f, a_o = 0.f;
    for (int k = 0; k < DH_; ++k) {
        float hv = sht[k];
        a_t += hv * tgW[k * DH_ + tid];
        a_a += hv * agW[k * DH_ + tid];
        a_h += hv * hgW[k * DH_ + tid];
        a_o += hv * outW[k * DH_ + tid];
    }
    for (int k = 0; k < DH_; ++k) {
        float xv = sx[k], qv = sq[k];
        a_t += xv * tgW[(DH_ + k) * DH_ + tid];
        a_a += xv * agW[(DH_ + k) * DH_ + tid];
        a_h += xv * hgW[(DH_ + k) * DH_ + tid];
        a_o += qv * outW[(DH_ + k) * DH_ + tid];
    }
    float tfv = tf[bt], afv = af[bt], hfv = hf[bt];
    float tg = sigm((a_t + tgb[tid]) * gatef(tfv));
    float ag = sigm((a_a + agb[tid]) * gatef(afv));
    float hg = sigm((a_h + hgb[tid]) * gatef(hfv));
    stg[tid] = tg; sag[tid] = ag; shg[tid] = hg;
    sred[tid] = sigm(a_o + outb[tid]);
    __syncthreads();
    for (int s = DH_ / 2; s > 0; s >>= 1) {
        if (tid < s) sred[tid] += sred[tid + s];
        __syncthreads();
    }
    if (tid == 0) out[b * L_ + t] = sred[0] / (float)DH_;

    // rank-R fusion -> learning gain
    float lgacc = 0.f;
    for (int r = 0; r < R_; ++r) {
        const float* twr = tw + (size_t)r * (DH_ + 1) * DH_;
        const float* awr = aw + (size_t)r * (DH_ + 1) * DH_;
        const float* hwr = hw + (size_t)r * (DH_ + 1) * DH_;
        float ft = twr[DH_ * DH_ + tid];   // pad row (i = DH)
        float fa = awr[DH_ * DH_ + tid];
        float fh = hwr[DH_ * DH_ + tid];
        for (int i = 0; i < DH_; ++i) {
            ft += stg[i] * twr[i * DH_ + tid];
            fa += sag[i] * awr[i * DH_ + tid];
            fh += shg[i] * hwr[i * DH_ + tid];
        }
        lgacc += Wff[r] * ft * fa * fh;
    }
    lg[b * DH_ + tid] = fmaxf(lgacc + cb[tid], 0.f);

    // C-broadcast part of forget logit: acts@W2 + f*colsums + bias
    float gb = g3b[tid] + tfv * s345[tid] + afv * s345[DH_ + tid] + hfv * s345[2 * DH_ + tid];
    for (int k = 0; k < DH_; ++k) gb += sx[k] * g3W[(DH_ + k) * DH_ + tid];
    g2b[b * DH_ + tid] = gb;
}

// h[b,c,:] = h * sigmoid(h @ W1 + g2bias[b,:]) + corr[b,c]*lg[b,:]   (WMMA GEMM)
// Also emits next step's read partials: part[b][ctile][h] = sum_c corr_{t+1}[b,c]*h_new[b,c,h]
__global__ void k_update(float* __restrict__ h, const _Float16* __restrict__ W1frag,
                         const float* __restrict__ g2b, const float* __restrict__ lg,
                         const float* __restrict__ qmat, const int* __restrict__ qseq,
                         float* __restrict__ part, int t) {
    __shared__ float tile[16 * DH_];                                   // raw fp32 h tile
    __shared__ __attribute__((aligned(32))) _Float16 afrag[4][32][16]; // A fragments (f16)
    __shared__ float scorr[16];                                        // corr at t
    __shared__ float scorr_n[16];                                      // corr at t+1
    int ctile = blockIdx.x, b = blockIdx.y;
    int tid = threadIdx.x;
    float* hb = h + ((size_t)b * C_ + ctile * 16) * DH_;

    // cooperative load: fp32 tile + f16 A-fragment swizzle
#pragma unroll
    for (int i = 0; i < 8; ++i) {
        int e = tid + i * 256;            // 0..2047 over 16 rows x 128 cols
        int row = e >> 7, K = e & 127;
        float v = hb[e];
        tile[e] = v;
        int kt = K >> 5, kk = K & 31;
        int hi = kk >> 4;                             // lane half
        int j  = kk & 15;                             // half index within fragment
        afrag[kt][(hi << 4) | row][j] = (_Float16)v;
    }
    if (tid < 16) {
        int qid = qseq[b * L_ + t];
        scorr[tid] = qmat[(size_t)qid * C_ + ctile * 16 + tid];
        if (t + 1 < L_) {
            int qidn = qseq[b * L_ + t + 1];
            scorr_n[tid] = qmat[(size_t)qidn * C_ + ctile * 16 + tid];
        }
    }
    __syncthreads();

    int wave = tid >> 5, lane = tid & 31;
    int noff = wave * 16;
    const v16h* Wf16 = (const v16h*)W1frag;

    v8f acc = {};
#pragma unroll
    for (int kt = 0; kt < 4; ++kt) {
        v16h av = *(const v16h*)&afrag[kt][lane][0];
        v16h bv = Wf16[(wave * 4 + kt) * 32 + lane];
        acc = __builtin_amdgcn_wmma_f32_16x16x32_f16(false, av, false, bv,
                                                     (short)0, acc, false, false);
    }

    const float* gbb = g2b + b * DH_;
    const float* lgb = lg + b * DH_;
    int N = noff + (lane & 15);
    int Mbase = (lane >> 4) << 3;
    float gN = gbb[N], lN = lgb[N];
    float hn[8];
#pragma unroll
    for (int i = 0; i < 8; ++i) {
        int M = Mbase + i;
        float forget = sigm(acc[i] + gN);
        hn[i] = tile[M * DH_ + N] * forget + scorr[M] * lN;
        hb[M * DH_ + N] = hn[i];
    }

    // next-step read partial: s = sum_M corr_{t+1}[M]*h_new[M,N]; fold lane<->lane+16
    if (t + 1 < L_) {
        float s = 0.f;
#pragma unroll
        for (int i = 0; i < 8; ++i) s += scorr_n[Mbase + i] * hn[i];
        s += __shfl_xor(s, 16, 32);
        if (lane < 16) part[((b * 16 + ctile) * DH_) + N] = s;
    }
}

// ---------------- launch ----------------

extern "C" void kernel_launch(void* const* d_in, const int* in_sizes, int n_in,
                              void* d_out, int out_size, void* d_ws, size_t ws_size,
                              hipStream_t stream) {
    const int*   qseq = (const int*)d_in[0];
    const int*   cseq = (const int*)d_in[1];
    const float* tf   = (const float*)d_in[2];
    const float* af   = (const float*)d_in[3];
    const float* hf   = (const float*)d_in[4];
    const float* qmat = (const float*)d_in[5];
    const float* eq   = (const float*)d_in[6];
    const float* ec   = (const float*)d_in[7];
    const float* inW  = (const float*)d_in[8];
    const float* inb  = (const float*)d_in[9];
    const float* inih = (const float*)d_in[10];
    const float* tgW  = (const float*)d_in[11];
    const float* tgb  = (const float*)d_in[12];
    const float* agW  = (const float*)d_in[13];
    const float* agb  = (const float*)d_in[14];
    const float* hgW  = (const float*)d_in[15];
    const float* hgb  = (const float*)d_in[16];
    const float* tw   = (const float*)d_in[17];
    const float* aw   = (const float*)d_in[18];
    const float* hw   = (const float*)d_in[19];
    const float* Wff  = (const float*)d_in[20];
    const float* cb   = (const float*)d_in[21];
    const float* g3W  = (const float*)d_in[22];
    const float* g3b  = (const float*)d_in[23];
    const float* outW = (const float*)d_in[24];
    const float* outb = (const float*)d_in[25];
    float* out = (float*)d_out;

    char* ws = (char*)d_ws;
    float* h      = (float*)ws;  ws += (size_t)B_ * C_ * DH_ * sizeof(float);
    float* acts   = (float*)ws;  ws += (size_t)B_ * L_ * DH_ * sizeof(float);
    float* part   = (float*)ws;  ws += (size_t)B_ * 16 * DH_ * sizeof(float);
    float* lgbuf  = (float*)ws;  ws += (size_t)B_ * DH_ * sizeof(float);
    float* g2b    = (float*)ws;  ws += (size_t)B_ * DH_ * sizeof(float);
    float* s345   = (float*)ws;  ws += 3 * DH_ * sizeof(float) + 512;  // pad for alignment
    _Float16* W1frag = (_Float16*)ws;

    k_init_h<<<C_ * DH_ / 256, 256, 0, stream>>>(h, inih);
    k_prep_w1<<<64, 256, 0, stream>>>(W1frag, s345, g3W);
    k_acts<<<B_ * L_, DH_, 0, stream>>>(acts, qseq, cseq, eq, ec, inW, inb);
    k_part0<<<dim3(16, B_), DH_, 0, stream>>>(part, inih, qmat, qseq);

    for (int t = 0; t < L_; ++t) {
        k_gates<<<B_, DH_, 0, stream>>>(lgbuf, g2b, out, part, acts, qseq, eq, tf, af, hf,
                                        tgW, tgb, agW, agb, hgW, hgb, tw, aw, hw,
                                        Wff, cb, g3W, g3b, s345, outW, outb, t);
        k_update<<<dim3(16, B_), 256, 0, stream>>>(h, W1frag, g2b, lgbuf, qmat, qseq, part, t);
    }
}
